// MultiHeadAttention_46136538694358
// MI455X (gfx1250) — compile-verified, matrix operand order checked by reading, not run on hardware
//
#include <hip/hip_runtime.h>
#include <hip/hip_bf16.h>

// MHA forward for B=2, S=2048, HID=1024, H=16, D=64 on gfx1250 (MI455X).
// GEMMs via v_wmma_f32_16x16x32_bf16; B-tiles staged with the Tensor Data
// Mover (tensor_load_to_lds + s_wait_tensorcnt), double-buffered in LDS.

#define B_ 2
#define S_ 2048
#define HID_ 1024
#define H_ 16
#define D_ 64
#define M_ (B_ * S_)   // 4096 rows for the projection GEMMs

typedef __attribute__((ext_vector_type(16))) __bf16 v16bf;
typedef __attribute__((ext_vector_type(8)))  float  v8f;
typedef __attribute__((ext_vector_type(4)))  unsigned int u32x4;
typedef __attribute__((ext_vector_type(8)))  int          i32x8;
typedef __attribute__((ext_vector_type(4)))  int          i32x4;

union FragU {
    uint4 q[2];
    v16bf v;
};

__device__ __forceinline__ unsigned short f2bf(float x) {
    unsigned int u = __builtin_bit_cast(unsigned int, x);
    u += 0x7FFFu + ((u >> 16) & 1u);   // round-to-nearest-even
    return (unsigned short)(u >> 16);
}

// Load a 16x32 bf16 A-fragment (or B-fragment via its transpose-equivalent
// layout) per the CDNA5 ISA layout: lane L -> row (L&15), the 16 per-lane
// elements are K = {g*8..g*8+7, 16+g*8..16+g*8+7} with g = L>>4.
// Both runs are contiguous 16-byte loads.
__device__ __forceinline__ v16bf load_frag(const unsigned short* base, int ld) {
    const int lane = threadIdx.x & 31;
    const int r = lane & 15;
    const int g = lane >> 4;
    FragU f;
    f.q[0] = *(const uint4*)(base + (size_t)r * ld + g * 8);
    f.q[1] = *(const uint4*)(base + (size_t)r * ld + 16 + g * 8);
    return f.v;
}

__device__ __forceinline__ v8f wmma_bf16(v16bf a, v16bf b, v8f c) {
    return __builtin_amdgcn_wmma_f32_16x16x32_bf16(
        /*neg_a=*/false, a, /*neg_b=*/false, b,
        /*c_mod=*/(short)0, c, /*reuse_a=*/false, /*reuse_b=*/false);
}

__device__ __forceinline__ float rmax16(float v) {
    #pragma unroll
    for (int m = 1; m < 16; m <<= 1) v = fmaxf(v, __shfl_xor(v, m, 32));
    return v;
}
__device__ __forceinline__ float rsum16(float v) {
    #pragma unroll
    for (int m = 1; m < 16; m <<= 1) v += __shfl_xor(v, m, 32);
    return v;
}

// Issue one TDM 2D-tile load: 64 rows x 32 cols of bf16 from src (row
// stride = ldK elements, tensor is rowsN x ldK) into LDS at byte offset
// lds_off.  Descriptor layout per CDNA5 ISA.
// This toolchain's builtin takes 6 args: (g0, g1, g2, g3, g_extra, cpol);
// groups 2/3 and the extra group are zero (disabled) for a 2D tensor.
__device__ __forceinline__ void tdm_load_tile_64x32(
    const unsigned short* src, unsigned int lds_off, int ldK, int rowsN) {
    const unsigned long long ga = (unsigned long long)(uintptr_t)src;
    u32x4 g0;
    g0[0] = 1u;                                   // count=1, no gather
    g0[1] = lds_off;                              // lds_addr (bytes)
    g0[2] = (unsigned int)ga;                     // global_addr[31:0]
    g0[3] = (unsigned int)((ga >> 32) & 0x1FFFFFFu) | (2u << 30);  // addr[56:32] | type=2
    i32x8 g1;
    g1[0] = 1 << 16;                              // data_size=1 (2 bytes)
    g1[1] = (ldK & 0xFFFF) << 16;                 // tensor_dim0[15:0]
    g1[2] = ((rowsN & 0xFFFF) << 16) | ((ldK >> 16) & 0xFFFF);  // tdim1 lo | tdim0 hi
    g1[3] = (32 << 16) | ((rowsN >> 16) & 0xFFFF);              // tile_dim0=32 | tdim1 hi
    g1[4] = 64;                                   // tile_dim1=64, tile_dim2=0
    g1[5] = ldK;                                  // tensor_dim0_stride[31:0]
    g1[6] = 0;
    g1[7] = 0;
    i32x4 z4 = {0, 0, 0, 0};
    i32x8 z8 = {0, 0, 0, 0, 0, 0, 0, 0};
    __builtin_amdgcn_tensor_load_to_lds(g0, g1, z4, z4, z8, 0);
}

// ---------------- Kernel 1: f32 -> bf16 elementwise ----------------
__global__ __launch_bounds__(256) void cvt_bf16_kernel(
    const float* __restrict__ in, unsigned short* __restrict__ out, int n) {
    int i = (blockIdx.x * 256 + threadIdx.x) * 4;
    if (i + 3 < n) {
        float4 v = *(const float4*)(in + i);
        out[i + 0] = f2bf(v.x);
        out[i + 1] = f2bf(v.y);
        out[i + 2] = f2bf(v.z);
        out[i + 3] = f2bf(v.w);
    }
}

// ------------- Kernel 2: transpose-convert W[K][N] f32 -> Wt[N][K] bf16 -------------
__global__ __launch_bounds__(256) void transpose_cvt_kernel(
    const float* __restrict__ in, unsigned short* __restrict__ out, int K, int N) {
    __shared__ float tile[32][33];
    const int bn = blockIdx.x * 32;   // N direction
    const int bk = blockIdx.y * 32;   // K direction
    const int tx = threadIdx.x & 31;
    const int ty = threadIdx.x >> 5;  // 0..7
    #pragma unroll
    for (int i = 0; i < 32; i += 8)
        tile[ty + i][tx] = in[(size_t)(bk + ty + i) * N + bn + tx];
    __syncthreads();
    #pragma unroll
    for (int i = 0; i < 32; i += 8)
        out[(size_t)(bn + ty + i) * K + bk + tx] = f2bf(tile[tx][ty + i]);
}

// ------------- Kernel 3: bf16 GEMM  C = A[MxK] @ Bt[NxK]^T + bias -------------
// B-tile (64x32) staged by the Tensor Data Mover, double-buffered.
// MODE 0: store bf16, head-split   out[((b*16+h)*2048+s)*64+d]      (q,k)
// MODE 1: store bf16, head-split-T out[((b*16+h)*64+d)*2048+s]      (v)
// MODE 2: store f32 row-major      out[m*N+n]                       (final)
template <int MODE>
__global__ __launch_bounds__(256) void gemm_bf16_kernel(
    const unsigned short* __restrict__ A, const unsigned short* __restrict__ Bt,
    const float* __restrict__ bias, void* __restrict__ out, int K, int N) {
    __shared__ __align__(16) unsigned short tileB[2][64 * 32];
    const int w    = threadIdx.x >> 5;
    const int lane = threadIdx.x & 31;
    const int r = lane & 15;
    const int g = lane >> 4;
    const int m0 = blockIdx.x * 128 + w * 16;
    const int n0 = blockIdx.y * 64;

    // LDS byte offsets of the two buffers (generic ptr low 32 bits == LDS addr)
    const unsigned int ldsOff0 = (unsigned int)(uintptr_t)&tileB[0][0];
    const unsigned int ldsOff1 = (unsigned int)(uintptr_t)&tileB[1][0];

    v8f acc[4] = {v8f{}, v8f{}, v8f{}, v8f{}};

    // Prologue: wave 0 kicks off the DMA for the first B tile.
    if (w == 0)
        tdm_load_tile_64x32(Bt + (size_t)n0 * K, ldsOff0, K, N);

    for (int kb = 0; kb < K; kb += 32) {
        const int cur = (kb >> 5) & 1;
        if (w == 0) {
            if (kb + 32 < K) {
                // issue next tile's DMA, then retire the current one
                tdm_load_tile_64x32(Bt + (size_t)n0 * K + kb + 32,
                                    cur ? ldsOff0 : ldsOff1, K, N);
                __builtin_amdgcn_s_wait_tensorcnt((short)1);
            } else {
                __builtin_amdgcn_s_wait_tensorcnt((short)0);
            }
        }
        __syncthreads();

        if (kb + 32 < K)
            __builtin_prefetch(A + (size_t)m0 * K + kb + 32 + (size_t)r * K, 0, 0);
        v16bf a = load_frag(A + (size_t)m0 * K + kb, K);
        #pragma unroll
        for (int t = 0; t < 4; ++t) {
            v16bf b = load_frag(&tileB[cur][(t * 16) * 32], 32);
            acc[t] = wmma_bf16(a, b, acc[t]);
        }
        __syncthreads();   // protect buffer before the next DMA overwrites it
    }

    #pragma unroll
    for (int t = 0; t < 4; ++t) {
        const int gn = n0 + t * 16 + r;
        const float bv = bias[gn];
        #pragma unroll
        for (int e = 0; e < 8; ++e) {
            const int gm = m0 + e + 8 * g;   // C layout: VGPR e -> row e + 8*(lane>=16)
            const float val = acc[t][e] + bv;
            if (MODE == 2) {
                ((float*)out)[(size_t)gm * N + gn] = val;
            } else {
                const unsigned short hv = f2bf(val);
                const int bb = gm >> 11, s = gm & 2047;
                const int hh = gn >> 6,  d = gn & 63;
                if (MODE == 0)
                    ((unsigned short*)out)[(((size_t)bb * H_ + hh) * S_ + s) * D_ + d] = hv;
                else
                    ((unsigned short*)out)[(((size_t)bb * H_ + hh) * D_ + d) * S_ + s] = hv;
            }
        }
    }
}

// ------------- Kernel 4: flash attention over one (b,h), 64 q rows/block -------------
// q,k: bf16 [BH][S][D]; vT: bf16 [BH][D][S]; ctx out: bf16 [B*S][HID] row-major.
__global__ __launch_bounds__(128) void attn_kernel(
    const unsigned short* __restrict__ qg, const unsigned short* __restrict__ kg,
    const unsigned short* __restrict__ vTg, unsigned short* __restrict__ ctx) {
    __shared__ __align__(16) unsigned short pbuf[4][16 * 32];
    const int w    = threadIdx.x >> 5;
    const int lane = threadIdx.x & 31;
    const int r = lane & 15;
    const int g = lane >> 4;
    const int bh = blockIdx.y;                 // 0..31
    const int m0 = blockIdx.x * 64 + w * 16;   // query tile within S

    const unsigned short* q  = qg  + (size_t)bh * S_ * D_;
    const unsigned short* k  = kg  + (size_t)bh * S_ * D_;
    const unsigned short* vT = vTg + (size_t)bh * D_ * S_;

    const v16bf qa0 = load_frag(q + (size_t)m0 * D_, D_);
    const v16bf qa1 = load_frag(q + (size_t)m0 * D_ + 32, D_);

    v8f acc[4] = {v8f{}, v8f{}, v8f{}, v8f{}};
    float mrun[8], lrun[8];
    #pragma unroll
    for (int e = 0; e < 8; ++e) { mrun[e] = -3.0e38f; lrun[e] = 0.0f; }
    const float scale = 0.125f;   // 1/sqrt(64)
    unsigned short* pl = pbuf[w];

    for (int kt = 0; kt < S_; kt += 32) {
        // scores: 16 q rows x 32 keys, K-dim = D = 64 (2 wmma per 16-key tile)
        v8f s0 = v8f{}, s1 = v8f{};
        s0 = wmma_bf16(qa0, load_frag(k + (size_t)kt * D_, D_), s0);
        s0 = wmma_bf16(qa1, load_frag(k + (size_t)kt * D_ + 32, D_), s0);
        s1 = wmma_bf16(qa0, load_frag(k + (size_t)(kt + 16) * D_, D_), s1);
        s1 = wmma_bf16(qa1, load_frag(k + (size_t)(kt + 16) * D_ + 32, D_), s1);

        // online softmax, rows e+8g, columns = lanes within 16-group
        #pragma unroll
        for (int e = 0; e < 8; ++e) {
            const float a0 = s0[e] * scale;
            const float a1 = s1[e] * scale;
            const float mx   = rmax16(fmaxf(a0, a1));
            const float mnew = fmaxf(mrun[e], mx);
            const float alpha = __expf(mrun[e] - mnew);
            const float p0 = __expf(a0 - mnew);
            const float p1 = __expf(a1 - mnew);
            lrun[e] = lrun[e] * alpha + rsum16(p0 + p1);
            mrun[e] = mnew;
            #pragma unroll
            for (int t = 0; t < 4; ++t) acc[t][e] *= alpha;
            const int row = e + 8 * g;
            pl[row * 32 + r]      = f2bf(p0);
            pl[row * 32 + 16 + r] = f2bf(p1);
        }
        // repack P (16x32) through per-wave LDS into A-fragment layout
        const v16bf pa = load_frag(pl, 32);
        #pragma unroll
        for (int t = 0; t < 4; ++t)
            acc[t] = wmma_bf16(pa, load_frag(vT + (size_t)(t * 16) * S_ + kt, S_), acc[t]);
    }

    const int bb = bh >> 4, hh = bh & 15;
    #pragma unroll
    for (int t = 0; t < 4; ++t) {
        #pragma unroll
        for (int e = 0; e < 8; ++e) {
            const int grow = bb * S_ + m0 + e + 8 * g;
            const int gcol = hh * D_ + t * 16 + r;
            ctx[(size_t)grow * HID_ + gcol] = f2bf(acc[t][e] / lrun[e]);
        }
    }
}

// ---------------------------------------------------------------------------
extern "C" void kernel_launch(void* const* d_in, const int* in_sizes, int n_in,
                              void* d_out, int out_size, void* d_ws, size_t ws_size,
                              hipStream_t stream) {
    (void)in_sizes; (void)n_in; (void)out_size; (void)ws_size;
    const float* Q  = (const float*)d_in[0];
    const float* K  = (const float*)d_in[1];
    const float* V  = (const float*)d_in[2];
    const float* Wq = (const float*)d_in[3];
    const float* bq = (const float*)d_in[4];
    const float* Wk = (const float*)d_in[5];
    const float* bk = (const float*)d_in[6];
    const float* Wv = (const float*)d_in[7];
    const float* bv = (const float*)d_in[8];
    const float* Wo = (const float*)d_in[9];
    const float* bo = (const float*)d_in[10];

    char* ws = (char*)d_ws;
    const size_t ACT = (size_t)M_ * HID_ * 2;   // 8 MiB bf16 activation
    const size_t WT  = (size_t)HID_ * HID_ * 2; // 2 MiB bf16 weight
    unsigned short* Qbf  = (unsigned short*)(ws);
    unsigned short* Kbf  = (unsigned short*)(ws + ACT);
    unsigned short* Vbf  = (unsigned short*)(ws + 2 * ACT);
    unsigned short* WqT  = (unsigned short*)(ws + 3 * ACT);
    unsigned short* WkT  = (unsigned short*)(ws + 3 * ACT + WT);
    unsigned short* WvT  = (unsigned short*)(ws + 3 * ACT + 2 * WT);
    unsigned short* WoT  = (unsigned short*)(ws + 3 * ACT + 3 * WT);
    unsigned short* qws  = (unsigned short*)(ws + 3 * ACT + 4 * WT);
    unsigned short* kws  = (unsigned short*)(ws + 4 * ACT + 4 * WT);
    unsigned short* vTws = (unsigned short*)(ws + 5 * ACT + 4 * WT);
    unsigned short* ctxw = (unsigned short*)(ws + 6 * ACT + 4 * WT);
    // total workspace use: 7*8MiB + 4*2MiB = 64 MiB

    const int nAct = M_ * HID_;
    dim3 cvtGrid(nAct / (256 * 4));
    cvt_bf16_kernel<<<cvtGrid, 256, 0, stream>>>(Q, Qbf, nAct);
    cvt_bf16_kernel<<<cvtGrid, 256, 0, stream>>>(K, Kbf, nAct);
    cvt_bf16_kernel<<<cvtGrid, 256, 0, stream>>>(V, Vbf, nAct);

    dim3 tGrid(HID_ / 32, HID_ / 32);
    transpose_cvt_kernel<<<tGrid, 256, 0, stream>>>(Wq, WqT, HID_, HID_);
    transpose_cvt_kernel<<<tGrid, 256, 0, stream>>>(Wk, WkT, HID_, HID_);
    transpose_cvt_kernel<<<tGrid, 256, 0, stream>>>(Wv, WvT, HID_, HID_);
    transpose_cvt_kernel<<<tGrid, 256, 0, stream>>>(Wo, WoT, HID_, HID_);

    dim3 gGrid(M_ / 128, HID_ / 64);
    gemm_bf16_kernel<0><<<gGrid, 256, 0, stream>>>(Qbf, WqT, bq, qws,  HID_, HID_);
    gemm_bf16_kernel<0><<<gGrid, 256, 0, stream>>>(Kbf, WkT, bk, kws,  HID_, HID_);
    gemm_bf16_kernel<1><<<gGrid, 256, 0, stream>>>(Vbf, WvT, bv, vTws, HID_, HID_);

    dim3 aGrid(S_ / 64, B_ * H_);
    attn_kernel<<<aGrid, 128, 0, stream>>>(qws, kws, vTws, ctxw);

    gemm_bf16_kernel<2><<<gGrid, 256, 0, stream>>>(ctxw, WoT, bo, d_out, HID_, HID_);
}
